// Attention_CD_18416819766012
// MI455X (gfx1250) — matrix operand order repaired and checked by reading, not compiled, not run on hardware
//
#include <hip/hip_runtime.h>
#include <hip/hip_bf16.h>
#include <stdint.h>

// ---------------------------------------------------------------------------
// MI455X (gfx1250) implementation. wave32, WMMA bf16 (f32 accumulate).
// softmax(1 - s*SCALE) == softmax(-s*SCALE)  ->  fold -SCALE into Q.
// Workspace layout (bf16 = uint16_t elements, QS = 4*16*1024*64 = 4194304):
//   [0*QS) q1  [1*QS) k1  [2*QS) v1^T   (v stored [B,H,D,L])
//   [3*QS) q2  [4*QS) k2  [5*QS) v2^T
//   [6*QS) xo  [7*QS) yo               ([B,L,H*D])
// Total 8*QS*2 bytes = 64 MB.
// All kernels: double-buffered LDS; GEMMs prefetch the next K-tile into
// registers during WMMA (pinned early with sched_barrier); pure-bf16 tile
// copies use GLOBAL_LOAD_ASYNC_TO_LDS (verified present on this toolchain).
// ---------------------------------------------------------------------------

#define DEV static __device__ __forceinline__

typedef __attribute__((ext_vector_type(16))) __bf16    v16bf;
typedef __attribute__((ext_vector_type(8)))  float     v8f;
typedef __attribute__((ext_vector_type(4)))  uint32_t  u32x4;
typedef __attribute__((ext_vector_type(4)))  int       i32x4;
typedef __attribute__((ext_vector_type(4)))  float     f32x4;

union Frag {
    v16bf bf;
    u32x4 q[2];
};

DEV uint32_t bf16_rn(float f) {
    union { float f; uint32_t u; } v; v.f = f;
    return (v.u + 0x7FFFu + ((v.u >> 16) & 1u)) >> 16;
}
DEV uint32_t pack2(float lo, float hi) {
    return (bf16_rn(hi) << 16) | (bf16_rn(lo) & 0xFFFFu);
}

DEV v8f wmma_bf16(const Frag& a, const Frag& b, v8f c) {
    return __builtin_amdgcn_wmma_f32_16x16x32_bf16(
        false, a.bf, false, b.bf, (short)0, c, false, false);
}

// A-fragment, 16x32 bf16 tile, row-major LDS [16][stride].
DEV void load_afrag(Frag& fr, const uint16_t* base, int lane, int stride) {
    const uint16_t* p = base + (lane & 15) * stride + ((lane >> 4) << 4);
    fr.q[0] = *(const u32x4*)p;          // K = hi*16 + 0..7
    fr.q[1] = *(const u32x4*)(p + 8);    // K = hi*16 + 8..15
}
// B-fragment, 32x16 bf16 tile stored N-major in LDS [16 n][stride k].
DEV void load_bfrag(Frag& fr, const uint16_t* base, int lane, int stride) {
    const uint16_t* p = base + (lane & 15) * stride + ((lane >> 4) << 4);
    fr.q[0] = *(const u32x4*)p;          // K = hi*16 + 0..7
    fr.q[1] = *(const u32x4*)(p + 8);    // K = hi*16 + 8..15
}

// ---- async global->LDS (CDNA5), guarded with synchronous fallback ---------
#if defined(__AMDGCN__) && __has_builtin(__builtin_amdgcn_global_load_async_to_lds_b128)
#define ASYNC_LDS 1
#else
#define ASYNC_LDS 0
#endif

DEV void cp16(uint16_t* lds, const uint16_t* g) {
#if ASYNC_LDS
    __builtin_amdgcn_global_load_async_to_lds_b128(
        (__attribute__((address_space(1))) i32x4*)g,
        (__attribute__((address_space(3))) i32x4*)lds, 0, 0);
#else
    *(u32x4*)lds = *(const u32x4*)g;
#endif
}
DEV void async_fence() {
#if ASYNC_LDS
#if __has_builtin(__builtin_amdgcn_s_wait_asynccnt)
    __builtin_amdgcn_s_wait_asynccnt(0);
#else
    asm volatile("s_wait_asynccnt 0" ::: "memory");
#endif
#endif
}
// Pin already-issued memory ops above subsequent compute (scheduler fence).
DEV void sched_fence() {
#if defined(__AMDGCN__) && __has_builtin(__builtin_amdgcn_sched_barrier)
    __builtin_amdgcn_sched_barrier(0);
#endif
}

static constexpr size_t QS = 4u * 16u * 1024u * 64u;   // 4194304 elements

// ---------------------------------------------------------------------------
// Kernel 1: qkv = x @ w_qkv  (M=4096, N=3072, K=1024), f32 in -> bf16 split out
// Double-buffered LDS, next K-tile prefetched into registers during WMMA.
// ---------------------------------------------------------------------------
__global__ __launch_bounds__(256) void qkv_gemm_kernel(
    const float* __restrict__ x, const float* __restrict__ y,
    const float* __restrict__ w1, const float* __restrict__ w2,
    uint16_t* __restrict__ ws)
{
    constexpr int K = 1024, N = 3072;
    __shared__ __align__(16) uint16_t As[2][128][40];
    __shared__ __align__(16) uint16_t Bs[2][128][40];

    const int tid  = threadIdx.x;
    const int lane = tid & 31;
    const int wave = tid >> 5;
    const int mw = (wave & 1) * 64;
    const int nw = (wave >> 1) * 32;
    const int mBase = blockIdx.x * 128;
    const int nBase = blockIdx.y * 128;
    const int z = blockIdx.z;

    const float* A = z ? y  : x;
    const float* W = z ? w2 : w1;
    uint16_t* qd  = ws + (size_t)z * 3u * QS;
    uint16_t* kd  = qd + QS;
    uint16_t* vtd = qd + 2u * QS;

    v8f acc[4][2];
    #pragma unroll
    for (int i = 0; i < 4; ++i) { acc[i][0] = {}; acc[i][1] = {}; }

    const int arow = tid >> 1, ac0 = (tid & 1) * 16;
    const int bk   = tid >> 3, bn0 = (tid & 7) * 16;
    const float* aptr = A + (size_t)(mBase + arow) * K + ac0;
    const float* bptr = W + (size_t)bk * N + nBase + bn0;

    f32x4 ar[4], br[4];
    auto g_load = [&](int k0) {
        const float* s = aptr + k0;
        ar[0] = *(const f32x4*)(s + 0);
        ar[1] = *(const f32x4*)(s + 4);
        ar[2] = *(const f32x4*)(s + 8);
        ar[3] = *(const f32x4*)(s + 12);
        const float* t = bptr + (size_t)k0 * N;
        #pragma unroll
        for (int g = 0; g < 4; ++g) br[g] = *(const f32x4*)(t + g * 4);
    };
    auto l_store = [&](int buf) {
        u32x4 p0 = { pack2(ar[0].x, ar[0].y), pack2(ar[0].z, ar[0].w),
                     pack2(ar[1].x, ar[1].y), pack2(ar[1].z, ar[1].w) };
        u32x4 p1 = { pack2(ar[2].x, ar[2].y), pack2(ar[2].z, ar[2].w),
                     pack2(ar[3].x, ar[3].y), pack2(ar[3].z, ar[3].w) };
        *(u32x4*)(&As[buf][arow][ac0])     = p0;
        *(u32x4*)(&As[buf][arow][ac0 + 8]) = p1;
        #pragma unroll
        for (int g = 0; g < 4; ++g) {            // transpose to Bs[n][k]
            Bs[buf][bn0 + g*4 + 0][bk] = (uint16_t)bf16_rn(br[g].x);
            Bs[buf][bn0 + g*4 + 1][bk] = (uint16_t)bf16_rn(br[g].y);
            Bs[buf][bn0 + g*4 + 2][bk] = (uint16_t)bf16_rn(br[g].z);
            Bs[buf][bn0 + g*4 + 3][bk] = (uint16_t)bf16_rn(br[g].w);
        }
    };

    g_load(0);
    l_store(0);
    __syncthreads();
    int buf = 0;
    for (int k0 = 0; k0 < K; k0 += 32) {
        int kn = (k0 + 32 < K) ? k0 + 32 : k0;   // clamped (last iter harmless)
        g_load(kn);                               // global prefetch ...
        sched_fence();                            // ... issued BEFORE the WMMAs
        Frag b0, b1, a[4];
        load_bfrag(b0, &Bs[buf][nw + 0][0],  lane, 40);
        load_bfrag(b1, &Bs[buf][nw + 16][0], lane, 40);
        #pragma unroll
        for (int i = 0; i < 4; ++i)
            load_afrag(a[i], &As[buf][mw + i * 16][0], lane, 40);
        #pragma unroll
        for (int i = 0; i < 4; ++i) {
            acc[i][0] = wmma_bf16(a[i], b0, acc[i][0]);
            acc[i][1] = wmma_bf16(a[i], b1, acc[i][1]);
        }
        l_store(buf ^ 1);
        __syncthreads();
        buf ^= 1;
    }

    const float NEGSCALE = -0.125f;    // -D^-0.5, D=64; folds softmax(1-s) sign
    const int rbase = (lane >> 4) * 8;
    #pragma unroll
    for (int i = 0; i < 4; ++i) {
        #pragma unroll
        for (int j = 0; j < 2; ++j) {
            #pragma unroll
            for (int v = 0; v < 8; ++v) {
                int m = mBase + mw + i * 16 + rbase + v;
                int n = nBase + nw + j * 16 + (lane & 15);
                int b = m >> 10, l = m & 1023;
                int sec = n >> 10, r = n & 1023;
                int h = r >> 6, d = r & 63;
                float val = acc[i][j][v];
                size_t bh = (size_t)(b * 16 + h);
                if (sec == 0)
                    qd[(bh * 1024u + l) * 64u + d] = (uint16_t)bf16_rn(val * NEGSCALE);
                else if (sec == 1)
                    kd[(bh * 1024u + l) * 64u + d] = (uint16_t)bf16_rn(val);
                else
                    vtd[(bh * 64u + d) * 1024u + l] = (uint16_t)bf16_rn(val);
            }
        }
    }
}

// ---------------------------------------------------------------------------
// Kernel 2: flash-style attention. 4 waves x 16 q-rows, key tiles of 32.
// Scores computed transposed (S^T = K * Q^T) so each lane owns one q column.
// K/V tiles double-buffered and staged with async global->LDS copies.
// ---------------------------------------------------------------------------
__global__ __launch_bounds__(128) void attn_kernel(uint16_t* __restrict__ ws)
{
    constexpr int L = 1024, D = 64, NT = L / 32;
    __shared__ __align__(16) uint16_t Ks[2][32][72];   // K tile [key][d]
    __shared__ __align__(16) uint16_t Vs[2][64][40];   // V^T tile [d][key]
    __shared__ __align__(16) uint16_t Ps[4][16][40];   // per-wave P [q][key]
    __shared__ float red[4][16];

    const int tid  = threadIdx.x;
    const int lane = tid & 31;
    const int wave = tid >> 5;
    const int branch = blockIdx.z & 1;
    const int b = blockIdx.z >> 1;
    const int h = blockIdx.y;
    const int qbase = blockIdx.x * 64 + wave * 16;

    // branch 0: q1 vs k2, values v1 ; branch 1: q2 vs k1, values v2
    const uint16_t* qp  = ws + (size_t)branch * 3u * QS;
    const uint16_t* kp  = ws + (size_t)(1 - branch) * 3u * QS + QS;
    const uint16_t* vtp = ws + (size_t)branch * 3u * QS + 2u * QS;
    uint16_t* outp = ws + 6u * QS + (size_t)branch * QS;

    const size_t bh = (size_t)(b * 16 + h);
    const uint16_t* qb  = qp  + bh * L * D;
    const uint16_t* kb  = kp  + bh * L * D;
    const uint16_t* vtb = vtp + bh * D * L;

    // staging: 2x 16B per thread per tile
    const int krow = tid >> 3, kseg = (tid & 7) * 8;
    const int vrow = tid >> 2, vseg = (tid & 3) * 8;
    auto stage = [&](int buf, int kt) {
        const uint16_t* ksrc = kb + (size_t)kt * 32 * D;
        cp16(&Ks[buf][krow][kseg],       ksrc + krow * D + kseg);
        cp16(&Ks[buf][krow + 16][kseg],  ksrc + (krow + 16) * D + kseg);
        const uint16_t* vsrc = vtb + (size_t)kt * 32;
        cp16(&Vs[buf][vrow][vseg],       vsrc + (size_t)vrow * L + vseg);
        cp16(&Vs[buf][vrow + 32][vseg],  vsrc + (size_t)(vrow + 32) * L + vseg);
    };

    // Q^T B-fragments (K-dim = d), loaded once straight from global
    Frag qf[2];
    {
        const uint16_t* qrow = qb + (size_t)(qbase + (lane & 15)) * D;
        int hi16 = (lane >> 4) * 16;
        #pragma unroll
        for (int f = 0; f < 2; ++f) {
            const uint16_t* p = qrow + f * 32 + hi16;
            qf[f].q[0] = *(const u32x4*)p;
            qf[f].q[1] = *(const u32x4*)(p + 8);
        }
    }

    v8f o[4];
    #pragma unroll
    for (int i = 0; i < 4; ++i) o[i] = {};
    float m_run = -1e30f, l_run = 0.0f;
    const int rbase = (lane >> 4) * 8;
    const int q = lane & 15;

    stage(0, 0);
    async_fence();
    __syncthreads();
    int buf = 0;

    for (int kt = 0; kt < NT; ++kt) {
        int ktn = (kt + 1 < NT) ? kt + 1 : kt;
        stage(buf ^ 1, ktn);               // async copies overlap compute below
        sched_fence();

        // S^T (keys x q), two 16x16 tiles, contraction over d (2x K=32)
        v8f s0 = {}, s1 = {};
        {
            Frag k0a, k0b, k1a, k1b;
            load_afrag(k0a, &Ks[buf][0][0],   lane, 72);
            load_afrag(k0b, &Ks[buf][0][32],  lane, 72);
            load_afrag(k1a, &Ks[buf][16][0],  lane, 72);
            load_afrag(k1b, &Ks[buf][16][32], lane, 72);
            s0 = wmma_bf16(k0a, qf[0], s0);
            s0 = wmma_bf16(k0b, qf[1], s0);
            s1 = wmma_bf16(k1a, qf[0], s1);
            s1 = wmma_bf16(k1b, qf[1], s1);
        }

        // online softmax; lane pair (l, l^16) together holds all 32 keys of q
        float vals[16];
        #pragma unroll
        for (int v = 0; v < 8; ++v) { vals[v] = s0[v]; vals[8 + v] = s1[v]; }
        float mloc = vals[0];
        #pragma unroll
        for (int v = 1; v < 16; ++v) mloc = fmaxf(mloc, vals[v]);
        mloc = fmaxf(mloc, __shfl_xor(mloc, 16, 32));
        float mnew = fmaxf(m_run, mloc);
        float corr = __expf(m_run - mnew);
        float pv[16];
        float psum = 0.0f;
        #pragma unroll
        for (int v = 0; v < 16; ++v) { pv[v] = __expf(vals[v] - mnew); psum += pv[v]; }
        psum += __shfl_xor(psum, 16, 32);
        l_run = l_run * corr + psum;
        m_run = mnew;

        if (lane < 16) red[wave][lane] = corr;        // per-q correction
        #pragma unroll
        for (int v = 0; v < 8; ++v) {                 // rescale O accumulators
            float c = red[wave][rbase + v];
            o[0][v] *= c; o[1][v] *= c; o[2][v] *= c; o[3][v] *= c;
        }

        // P^T registers -> LDS as P[q][key] (bf16), becomes an A-fragment
        #pragma unroll
        for (int v = 0; v < 8; ++v) {
            Ps[wave][q][rbase + v]      = (uint16_t)bf16_rn(pv[v]);
            Ps[wave][q][16 + rbase + v] = (uint16_t)bf16_rn(pv[8 + v]);
        }
        Frag pa, vb[4];
        load_afrag(pa, &Ps[wave][0][0], lane, 40);
        #pragma unroll
        for (int dt = 0; dt < 4; ++dt)
            load_bfrag(vb[dt], &Vs[buf][dt * 16][0], lane, 40);
        #pragma unroll
        for (int dt = 0; dt < 4; ++dt)                // O += P @ V
            o[dt] = wmma_bf16(pa, vb[dt], o[dt]);

        async_fence();
        __syncthreads();
        buf ^= 1;
    }

    if (lane < 16) red[wave][lane] = 1.0f / l_run;
    #pragma unroll
    for (int dt = 0; dt < 4; ++dt) {
        int d = dt * 16 + q;
        #pragma unroll
        for (int v = 0; v < 8; ++v) {
            float inv = red[wave][rbase + v];
            int lg = qbase + rbase + v;
            outp[(((size_t)b * 1024u + lg) * 16u + h) * 64u + d] =
                (uint16_t)bf16_rn(o[dt][v] * inv);
        }
    }
}

// ---------------------------------------------------------------------------
// Kernel 3: out = xo @ w_p + b  (M=4096, N=256, K=1024), bf16 A, f32 out
// A tiles staged with async global->LDS; weights converted via registers.
// ---------------------------------------------------------------------------
__global__ __launch_bounds__(256) void proj_gemm_kernel(
    const uint16_t* __restrict__ xo, const float* __restrict__ wp1,
    const float* __restrict__ wp2, const float* __restrict__ bp1,
    const float* __restrict__ bp2, float* __restrict__ out)
{
    constexpr int K = 1024, N = 256;
    __shared__ __align__(16) uint16_t As[2][128][40];
    __shared__ __align__(16) uint16_t Bs[2][128][40];

    const int tid = threadIdx.x, lane = tid & 31, wave = tid >> 5;
    const int mw = (wave & 1) * 64, nw = (wave >> 1) * 32;
    const int mBase = blockIdx.x * 128, nBase = blockIdx.y * 128;
    const int z = blockIdx.z;

    const uint16_t* A = xo + (size_t)z * QS;
    const float* W    = z ? wp2 : wp1;
    const float* bias = z ? bp2 : bp1;
    float* op = out + (size_t)z * (4u * 1024u * 256u);

    v8f acc[4][2];
    #pragma unroll
    for (int i = 0; i < 4; ++i) { acc[i][0] = {}; acc[i][1] = {}; }

    const int arow = tid >> 1, ac0 = (tid & 1) * 16;
    const int bk = tid >> 3, bn0 = (tid & 7) * 16;
    const uint16_t* asrc = A + (size_t)(mBase + arow) * K + ac0;
    const float*    bsrc = W + (size_t)bk * N + nBase + bn0;

    f32x4 br[4];
    auto stage_a = [&](int buf, int k0) {            // pure bf16 copy -> async
        cp16(&As[buf][arow][ac0],     asrc + k0);
        cp16(&As[buf][arow][ac0 + 8], asrc + k0 + 8);
    };
    auto g_load_b = [&](int k0) {
        const float* t = bsrc + (size_t)k0 * N;
        #pragma unroll
        for (int g = 0; g < 4; ++g) br[g] = *(const f32x4*)(t + g * 4);
    };
    auto l_store_b = [&](int buf) {
        #pragma unroll
        for (int g = 0; g < 4; ++g) {
            Bs[buf][bn0 + g*4 + 0][bk] = (uint16_t)bf16_rn(br[g].x);
            Bs[buf][bn0 + g*4 + 1][bk] = (uint16_t)bf16_rn(br[g].y);
            Bs[buf][bn0 + g*4 + 2][bk] = (uint16_t)bf16_rn(br[g].z);
            Bs[buf][bn0 + g*4 + 3][bk] = (uint16_t)bf16_rn(br[g].w);
        }
    };

    stage_a(0, 0);
    g_load_b(0);
    l_store_b(0);
    async_fence();
    __syncthreads();
    int buf = 0;

    for (int k0 = 0; k0 < K; k0 += 32) {
        int kn = (k0 + 32 < K) ? k0 + 32 : k0;
        stage_a(buf ^ 1, kn);                        // async, overlaps WMMAs
        g_load_b(kn);                                // f32 weight prefetch
        sched_fence();                               // keep issues above WMMAs
        Frag b0, b1, a[4];
        load_bfrag(b0, &Bs[buf][nw + 0][0],  lane, 40);
        load_bfrag(b1, &Bs[buf][nw + 16][0], lane, 40);
        #pragma unroll
        for (int i = 0; i < 4; ++i)
            load_afrag(a[i], &As[buf][mw + i * 16][0], lane, 40);
        #pragma unroll
        for (int i = 0; i < 4; ++i) {
            acc[i][0] = wmma_bf16(a[i], b0, acc[i][0]);
            acc[i][1] = wmma_bf16(a[i], b1, acc[i][1]);
        }
        l_store_b(buf ^ 1);
        async_fence();
        __syncthreads();
        buf ^= 1;
    }

    const int rbase = (lane >> 4) * 8;
    #pragma unroll
    for (int i = 0; i < 4; ++i) {
        #pragma unroll
        for (int j = 0; j < 2; ++j) {
            int n = nBase + nw + j * 16 + (lane & 15);
            float bn = bias[n];
            #pragma unroll
            for (int v = 0; v < 8; ++v) {
                int m = mBase + mw + i * 16 + rbase + v;
                op[(size_t)m * N + n] = acc[i][j][v] + bn;
            }
        }
    }
}

// ---------------------------------------------------------------------------
extern "C" void kernel_launch(void* const* d_in, const int* in_sizes, int n_in,
                              void* d_out, int out_size, void* d_ws, size_t ws_size,
                              hipStream_t stream) {
    (void)in_sizes; (void)n_in; (void)out_size; (void)ws_size;
    const float* x   = (const float*)d_in[0];
    const float* y   = (const float*)d_in[1];
    const float* w1  = (const float*)d_in[2];
    const float* w2  = (const float*)d_in[3];
    const float* wp1 = (const float*)d_in[4];
    const float* bp1 = (const float*)d_in[5];
    const float* wp2 = (const float*)d_in[6];
    const float* bp2 = (const float*)d_in[7];
    uint16_t* ws = (uint16_t*)d_ws;          // needs 64 MB
    float* out = (float*)d_out;

    qkv_gemm_kernel<<<dim3(32, 24, 2), 256, 0, stream>>>(x, y, w1, w2, ws);
    attn_kernel<<<dim3(16, 16, 8), 128, 0, stream>>>(ws);
    proj_gemm_kernel<<<dim3(32, 2, 2), 256, 0, stream>>>(
        ws + 6u * QS, wp1, wp2, bp1, bp2, out);
}
